// SRUCell_8924942042062
// MI455X (gfx1250) — matrix operand order repaired and checked by reading, not compiled, not run on hardware
//
#include <hip/hip_runtime.h>
#include <hip/hip_bf16.h>

typedef __bf16 bf16;
typedef __attribute__((ext_vector_type(16))) __bf16 v16bf;
typedef __attribute__((ext_vector_type(8)))  __bf16 v8bf;
typedef __attribute__((ext_vector_type(8)))  float  v8f;

#define L_DIM 2048
#define B_DIM 16
#define D_DIM 1024
#define N_DIM 3072               // 3*d
#define M_DIM (L_DIM * B_DIM)    // 32768
#define K_DIM D_DIM              // 1024

#define BM 128
#define BN 128
#define BK 32
#define NTILES (K_DIM / BK)      // 32
#define CHUNKS 16
#define CHLEN (L_DIM / CHUNKS)   // 128

// Async global->LDS copy, 16 bytes/lane, tracked by ASYNCcnt (CDNA5).
// INST_OFFSET is added to BOTH the LDS and global addresses (ISA 08 §4.4).
#define ASYNC_LD16(ldsaddr, voff, sbase, imm)                                  \
  asm volatile("global_load_async_to_lds_b128 %0, %1, %2 offset:" #imm        \
               :: "v"(ldsaddr), "v"(voff), "s"(sbase) : "memory")

// ---------------------------------------------------------------------------
// fp32 -> bf16 conversion of x (keeps [M,K] row-major layout)
// ---------------------------------------------------------------------------
__global__ void cvt_x_kernel(const float* __restrict__ x, bf16* __restrict__ x16) {
  long i = ((long)blockIdx.x * blockDim.x + threadIdx.x) * 4;
  float4 v = *(const float4*)(x + i);
  union { bf16 h[4]; uint2 u; } p;
  p.h[0] = (bf16)v.x; p.h[1] = (bf16)v.y; p.h[2] = (bf16)v.z; p.h[3] = (bf16)v.w;
  *(uint2*)(x16 + i) = p.u;
}

// ---------------------------------------------------------------------------
// fp32 weight [K,N] -> bf16 transposed [N,K] (coalesced reads; tiny 6 MB write)
// ---------------------------------------------------------------------------
__global__ void cvt_wt_kernel(const float* __restrict__ w, bf16* __restrict__ wt16) {
  long idx = (long)blockIdx.x * blockDim.x + threadIdx.x;   // over N_DIM*K_DIM
  int n = (int)(idx % N_DIM);
  int k = (int)(idx / N_DIM);
  wt16[(long)n * K_DIM + k] = (bf16)w[idx];
}

// ---------------------------------------------------------------------------
// GEMM: act = activation(x16 @ wt16^T + bias)
// Double-buffered LDS fed by GLOBAL_LOAD_ASYNC_TO_LDS_B128 (ASYNCcnt),
// 8x v_wmma_f32_16x16x32_bf16 per K-step per wave.
// ---------------------------------------------------------------------------
__global__ __launch_bounds__(256) void gemm_act_kernel(
    const bf16* __restrict__ x16, const bf16* __restrict__ wt16,
    const float* __restrict__ bias, float* __restrict__ act) {
  __shared__ __align__(32) bf16 As[2][BM * BK];   // [buf][row][k]
  __shared__ __align__(32) bf16 Bs[2][BN * BK];   // [buf][n][k] (K-transposed)

  const int tid   = threadIdx.x;
  const int lane  = tid & 31;
  const int wave  = tid >> 5;       // 0..7
  const int waveM = wave & 3;       // 4 wave rows
  const int waveN = wave >> 2;      // 2 wave cols
  const int m16   = lane & 15;
  const int hi    = lane >> 4;

  const int mBase = blockIdx.y * BM;
  const int nBase = blockIdx.x * BN;

  const int ldr     = tid >> 1;          // 0..127: row (A) / n (B)
  const int lds_seg = (tid & 1) * 16;    // 16-element half of the 32-wide K slab

  // LDS byte addresses (low 32 bits of the flat LDS-aperture address)
  unsigned ldsA[2], ldsB[2];
  ldsA[0] = (unsigned)(uintptr_t)&As[0][ldr * BK + lds_seg];
  ldsA[1] = (unsigned)(uintptr_t)&As[1][ldr * BK + lds_seg];
  ldsB[0] = (unsigned)(uintptr_t)&Bs[0][ldr * BK + lds_seg];
  ldsB[1] = (unsigned)(uintptr_t)&Bs[1][ldr * BK + lds_seg];

  // Per-thread global byte offsets (32-bit, GVS mode voffset)
  const unsigned aOff0 = (unsigned)(((mBase + ldr) * K_DIM + lds_seg) * 2);
  const unsigned bOff0 = (unsigned)(((nBase + ldr) * K_DIM + lds_seg) * 2);

  v8f acc[2][4];
  #pragma unroll
  for (int i = 0; i < 2; i++)
    #pragma unroll
    for (int j = 0; j < 4; j++)
      #pragma unroll
      for (int e = 0; e < 8; e++) acc[i][j][e] = 0.0f;

  // Prologue: stream tile 0 into buffer 0 (4 async ops per wave)
  {
    unsigned vA = aOff0, vB = bOff0;
    ASYNC_LD16(ldsA[0], vA, x16, 0);
    ASYNC_LD16(ldsA[0], vA, x16, 16);
    ASYNC_LD16(ldsB[0], vB, wt16, 0);
    ASYNC_LD16(ldsB[0], vB, wt16, 16);
  }

  for (int t = 0; t < NTILES; t++) {
    const int buf = t & 1;
    if (t + 1 < NTILES) {
      // Issue next tile into the other buffer; its target buffer was last
      // read in iteration t-1, whose trailing barrier has already passed.
      unsigned vA = aOff0 + (unsigned)((t + 1) * BK * 2);
      unsigned vB = bOff0 + (unsigned)((t + 1) * BK * 2);
      ASYNC_LD16(ldsA[buf ^ 1], vA, x16, 0);
      ASYNC_LD16(ldsA[buf ^ 1], vA, x16, 16);
      ASYNC_LD16(ldsB[buf ^ 1], vB, wt16, 0);
      ASYNC_LD16(ldsB[buf ^ 1], vB, wt16, 16);
      asm volatile("s_wait_asynccnt 0x4" ::: "memory");  // tile t complete
    } else {
      asm volatile("s_wait_asynccnt 0x0" ::: "memory");
    }
    __syncthreads();   // all waves' tile-t data visible in LDS

    // A fragments: 16-bit A 16x32 layout -> lane(m=m16): K = 8*hi+i / 16+8*hi+i
    v16bf afrag[2];
    #pragma unroll
    for (int fm = 0; fm < 2; fm++) {
      int row = waveM * 32 + fm * 16 + m16;
      v8bf a0 = *(const v8bf*)&As[buf][row * BK + 8 * hi];
      v8bf a1 = *(const v8bf*)&As[buf][row * BK + 16 + 8 * hi];
      #pragma unroll
      for (int i = 0; i < 8; i++) { afrag[fm][i] = a0[i]; afrag[fm][i + 8] = a1[i]; }
    }

    // B fragments: B 32x16 -> lane(n=m16): K = 8*hi+i / 16+8*hi+i (matches A)
    #pragma unroll
    for (int fn = 0; fn < 4; fn++) {
      int col = waveN * 64 + fn * 16 + m16;
      v8bf b0 = *(const v8bf*)&Bs[buf][col * BK + 8 * hi];
      v8bf b1 = *(const v8bf*)&Bs[buf][col * BK + 16 + 8 * hi];
      v16bf bfrag;
      #pragma unroll
      for (int i = 0; i < 8; i++) { bfrag[i] = b0[i]; bfrag[i + 8] = b1[i]; }
      #pragma unroll
      for (int fm = 0; fm < 2; fm++) {
        acc[fm][fn] = __builtin_amdgcn_wmma_f32_16x16x32_bf16(
            false, afrag[fm], false, bfrag, (short)0, acc[fm][fn], false, false);
      }
    }
    __syncthreads();   // reads of buf done before it is refilled at t+2
  }

  // Epilogue: bias + branch-free fast activations.
  //   sigmoid(v) = rcp(1 + exp(-v));  tanh(v) = 2*sigmoid(2v) - 1
  #pragma unroll
  for (int fn = 0; fn < 4; fn++) {
    int n_g = nBase + waveN * 64 + fn * 16 + m16;
    float bcol = bias[n_g];
    bool is_tanh = (n_g >> 10) == 0;     // region 0 = w0 -> tanh
    #pragma unroll
    for (int fm = 0; fm < 2; fm++) {
      int row0 = mBase + waveM * 32 + fm * 16 + 8 * hi;
      #pragma unroll
      for (int j = 0; j < 8; j++) {
        float v  = acc[fm][fn][j] + bcol;
        float vs = is_tanh ? 2.0f * v : v;
        float s  = __builtin_amdgcn_rcpf(1.0f + __expf(-vs));
        float o  = is_tanh ? 2.0f * s - 1.0f : s;
        act[(long)(row0 + j) * N_DIM + n_g] = o;
      }
    }
  }
}

// ---------------------------------------------------------------------------
// Chunked linear-recurrence scan.
// P1: per-chunk composition (A = prod f, S = running affine offset)
// ---------------------------------------------------------------------------
__global__ void scan_p1_kernel(const float* __restrict__ act,
                               float* __restrict__ Aout, float* __restrict__ Sout) {
  int id   = blockIdx.x * blockDim.x + threadIdx.x;  // < CHUNKS * B*d
  int ch   = id >> 14;        // / 16384
  int cidx = id & 16383;
  int b    = cidx >> 10;
  int dch  = cidx & 1023;
  float A = 1.0f, S = 0.0f;
  const float* p = act + (long)(ch * CHLEN * B_DIM + b) * N_DIM + dch;
  for (int l = 0; l < CHLEN; l++) {
    float tilde = p[0];
    float f     = p[D_DIM];
    S = f * S + (1.0f - f) * tilde;
    A *= f;
    p += (long)B_DIM * N_DIM;
  }
  Aout[id] = A;
  Sout[id] = S;
}

// P2: sequential combine over the 16 chunks per channel; emits carries + c_last
__global__ void scan_p2_kernel(const float* __restrict__ c0,
                               const float* __restrict__ Aarr,
                               const float* __restrict__ Sarr,
                               float* __restrict__ carry,
                               float* __restrict__ c_last) {
  int cidx = blockIdx.x * blockDim.x + threadIdx.x;  // < B*d
  float c = c0[cidx];
  #pragma unroll
  for (int ch = 0; ch < CHUNKS; ch++) {
    int idx = ch * (B_DIM * D_DIM) + cidx;
    carry[idx] = c;
    c = Aarr[idx] * c + Sarr[idx];
  }
  c_last[cidx] = c;
}

// P3: replay each chunk from its carry; write highway output h
__global__ void scan_p3_kernel(const float* __restrict__ act,
                               const float* __restrict__ x,
                               const float* __restrict__ carry,
                               float* __restrict__ h) {
  int id   = blockIdx.x * blockDim.x + threadIdx.x;
  int ch   = id >> 14;
  int cidx = id & 16383;
  int b    = cidx >> 10;
  int dch  = cidx & 1023;
  float c = carry[id];
  long row0 = (long)(ch * CHLEN) * B_DIM + b;
  const float* pa = act + row0 * N_DIM + dch;
  const float* px = x   + row0 * D_DIM + dch;
  float*       ph = h   + row0 * D_DIM + dch;
  for (int l = 0; l < CHLEN; l++) {
    float tilde = pa[0];
    float f     = pa[D_DIM];
    float r     = pa[2 * D_DIM];
    c = f * c + (1.0f - f) * tilde;
    float xv = px[0];
    ph[0] = c * r + xv * (1.0f - r);
    pa += (long)B_DIM * N_DIM;
    px += (long)B_DIM * D_DIM;
    ph += (long)B_DIM * D_DIM;
  }
}

// ---------------------------------------------------------------------------
extern "C" void kernel_launch(void* const* d_in, const int* in_sizes, int n_in,
                              void* d_out, int out_size, void* d_ws, size_t ws_size,
                              hipStream_t stream) {
  const float* x      = (const float*)d_in[0];
  const float* c0     = (const float*)d_in[1];
  const float* weight = (const float*)d_in[2];
  const float* bias   = (const float*)d_in[3];
  float* out = (float*)d_out;

  // Workspace layout
  char* ws = (char*)d_ws;
  const size_t ACT_BYTES = (size_t)M_DIM * N_DIM * sizeof(float);   // 384 MB
  const size_t X16_BYTES = (size_t)M_DIM * K_DIM * sizeof(bf16);    //  64 MB
  const size_t WT_BYTES  = (size_t)N_DIM * K_DIM * sizeof(bf16);    //   6 MB
  const size_t AS_BYTES  = (size_t)CHUNKS * B_DIM * D_DIM * sizeof(float); // 1 MB

  float* act   = (float*)(ws);
  bf16*  x16   = (bf16*)(ws + ACT_BYTES);
  bf16*  wt16  = (bf16*)(ws + ACT_BYTES + X16_BYTES);
  float* Aarr  = (float*)(ws + ACT_BYTES + X16_BYTES + WT_BYTES);
  float* Sarr  = (float*)((char*)Aarr + AS_BYTES);
  float* carry = (float*)((char*)Sarr + AS_BYTES);

  cvt_x_kernel<<<(int)(((long)M_DIM * K_DIM) / (256 * 4)), 256, 0, stream>>>(x, x16);
  cvt_wt_kernel<<<(int)(((long)N_DIM * K_DIM) / 256), 256, 0, stream>>>(weight, wt16);

  dim3 gGemm(N_DIM / BN, M_DIM / BM);   // (24, 256)
  gemm_act_kernel<<<gGemm, 256, 0, stream>>>(x16, wt16, bias, act);

  const int scanThreads = CHUNKS * B_DIM * D_DIM;   // 262144
  scan_p1_kernel<<<scanThreads / 256, 256, 0, stream>>>(act, Aarr, Sarr);
  scan_p2_kernel<<<(B_DIM * D_DIM) / 256, 256, 0, stream>>>(
      c0, Aarr, Sarr, carry, out + (size_t)M_DIM * D_DIM);
  scan_p3_kernel<<<scanThreads / 256, 256, 0, stream>>>(act, x, carry, out);
}
